// NCCAModel2_50818053046300
// MI455X (gfx1250) — matrix operand order repaired and checked by reading, not compiled, not run on hardware
//
#include <hip/hip_runtime.h>

typedef __attribute__((ext_vector_type(16))) _Float16 v16h;
typedef __attribute__((ext_vector_type(8)))  _Float16 v8h;
typedef __attribute__((ext_vector_type(8)))  float    v8f;

#define CH    16
#define TW    16
#define THT   8
#define XS_H  14            // THT + 6 halo
#define XS_W  22            // TW + 6 halo
#define XS_PS 20            // padded floats per pixel (80 B, bank-conflict free)
#define PERC  128
#define PERC_P 136          // padded row (272 B -> bank step 4)
#define HID   256
#define NPIX  128

// Dynamic LDS layout (bytes), 16B aligned except OFF_PREL (8B):
#define OFF_XS    0                               // 14*22*20*4  = 24640
#define OFF_PERC  24640                           // 128*136*2   = 34816
#define OFF_STAGE 59456                           // 128*40*2    = 10240
#define OFF_W0S   69696                           // 256*136*2   = 69632
#define OFF_KERN  139328                          // 6*49*4      = 1176
#define OFF_PREL  140504                          // 128
#define SMEM_BYTES 140640

__device__ __forceinline__ v16h cat8(v8h lo, v8h hi) {
  return __builtin_shufflevector(lo, hi, 0,1,2,3,4,5,6,7,8,9,10,11,12,13,14,15);
}
__device__ __forceinline__ unsigned lds_off(const void* p) {
  // generic shared address = {shared aperture, LDS offset[31:0]}
  return (unsigned)(unsigned long long)p;
}
__device__ __forceinline__ void async_g2l_b128(unsigned lds, const void* g) {
  asm volatile("global_load_async_to_lds_b128 %0, %1, off"
               :: "v"(lds), "v"((unsigned long long)g) : "memory");
}
__device__ __forceinline__ void wait_async0() {
  asm volatile("s_wait_asynccnt 0" ::: "memory");
}

// ---------------------------------------------------------------------------
__global__ __launch_bounds__(256) void nca_prep(
    const float* __restrict__ W0, const float* __restrict__ W1,
    _Float16* __restrict__ W0T, _Float16* __restrict__ W1T)
{
  int t = blockIdx.x * 256 + threadIdx.x;
  if (t < 128 * 256) {                       // W0 [128][256] -> W0T [256][128]
    int k = t >> 8, n = t & 255;
    W0T[n * 128 + k] = (_Float16)W0[t];
  }
  if (t < 256 * 16) {                        // W1 [256][16]  -> W1T [16][256]
    int k = t >> 4, n = t & 15;
    W1T[n * 256 + k] = (_Float16)W1[t];
  }
}

// ---------------------------------------------------------------------------
__global__ __launch_bounds__(256) void nca_update(
    const float* __restrict__ xin,           // [B,H,W,16]
    float* __restrict__ xmid,                // [B,H,W,16]
    unsigned char* __restrict__ preMask,     // [B,H,W]
    const float* __restrict__ stoch,         // [B,H,W,1] (this step)
    const float* __restrict__ b0,            // [256]
    const _Float16* __restrict__ W0T,        // [256][128]
    const _Float16* __restrict__ W1T,        // [16][256]
    int Hd, int Wd)
{
  extern __shared__ __align__(16) char smem[];
  float*         xsf   = (float*)(smem + OFF_XS);
  _Float16*      percA = (_Float16*)(smem + OFF_PERC);
  _Float16*      stg   = (_Float16*)(smem + OFF_STAGE);
  _Float16*      w0s   = (_Float16*)(smem + OFF_W0S);
  float*         kern  = (float*)(smem + OFF_KERN);
  unsigned char* preL  = (unsigned char*)(smem + OFF_PREL);

#define XS(y,x,c)  xsf[(((y)*XS_W)+(x))*XS_PS+(c)]
#define PA(p,k)    percA[(p)*PERC_P+(k)]
#define STG(w,m,k) stg[(((w)*16)+(m))*40+(k)]
#define W0S(n,k)   w0s[(n)*PERC_P+(k)]

  const int tid = threadIdx.x;
  const int b   = blockIdx.z;
  const int ty0 = blockIdx.y * THT;
  const int tx0 = blockIdx.x * TW;

  // ---- Issue async DMA: W0T -> LDS (padded rows), 4096 chunks of 16B ----
  {
    const unsigned base = lds_off(w0s);
    for (int i = tid; i < 4096; i += 256) {
      int n = i >> 4, j = i & 15;
      async_g2l_b128(base + (unsigned)(n * (PERC_P * 2) + j * 16),
                     (const char*)W0T + (size_t)n * 256 + j * 16);
    }
  }

  // ---- x tile + halo: async path for interior tiles, guarded otherwise ----
  const bool interior = (ty0 >= 3) && (ty0 + THT + 3 <= Hd) &&
                        (tx0 >= 3) && (tx0 + TW + 3 <= Wd);
  if (interior) {
    const unsigned base = lds_off(xsf);
    const char* rowbase =
        (const char*)(xin + (((size_t)b * Hd + (ty0 - 3)) * Wd + (tx0 - 3)) * CH);
    const size_t rowpitch = (size_t)Wd * CH * 4;
    for (int i = tid; i < XS_H * XS_W * 4; i += 256) {   // 4x16B per pixel
      int pix = i >> 2, q = i & 3;
      int yy = pix / XS_W, xx = pix % XS_W;
      async_g2l_b128(base + (unsigned)(((yy * XS_W + xx) * XS_PS + q * 4) * 4),
                     rowbase + (size_t)yy * rowpitch + (size_t)(xx * 4 + q) * 16);
    }
  } else {
    for (int i = tid; i < XS_H * XS_W; i += 256) {
      int yy = i / XS_W, xx = i % XS_W;
      int gy = ty0 + yy - 3, gx = tx0 + xx - 3;
      float4 v0 = {}, v1 = {}, v2 = {}, v3 = {};
      if (gy >= 0 && gy < Hd && gx >= 0 && gx < Wd) {
        const float4* src =
            (const float4*)(xin + (((size_t)b * Hd + gy) * Wd + gx) * CH);
        v0 = src[0]; v1 = src[1]; v2 = src[2]; v3 = src[3];
      }
      float4* dst = (float4*)&XS(yy, xx, 0);
      dst[0] = v0; dst[1] = v1; dst[2] = v2; dst[3] = v3;
    }
  }

  // ---- Sobel kernel bank (VALU work overlapping the async DMA) ----
  for (int t = tid; t < 6 * 49; t += 256) {
    int k = t / 49, pos = t % 49;
    int s = (3 + 2 * (k >> 1)) >> 1;                 // 1,1,2,2,3,3
    int py = pos / 7 - 3, px = pos % 7 - 3;
    float val = 0.f;
    if (py >= -s && py <= s && px >= -s && px <= s) {
      float den = (float)(px * px + py * py); if (den == 0.f) den = 1.f;
      val = ((k & 1) ? (float)py : (float)px) / den;
    }
    float norm = 0.f;
    for (int yy = -s; yy <= s; ++yy)
      for (int xx = -s; xx <= s; ++xx) {
        float dd = (float)(xx * xx + yy * yy); if (dd == 0.f) dd = 1.f;
        norm += fabsf(((k & 1) ? (float)yy : (float)xx) / dd);
      }
    kern[k * 49 + pos] = val / norm;
  }

  wait_async0();
  __syncthreads();

  // ---- Perception -> percA (f16, A-fragment friendly, padded rows) ----
  {
    const int p = tid & 127, half = tid >> 7;
    const int ly = p >> 4, lx = p & 15;
    const int c0 = half * 8;
    float acc[8][6]; float m5[8];
    for (int c = 0; c < 8; ++c) { m5[c] = -3.0e38f; for (int k = 0; k < 6; ++k) acc[c][k] = 0.f; }
    for (int dy = 0; dy < 7; ++dy)
      for (int dx = 0; dx < 7; ++dx) {
        const int ki = dy * 7 + dx;
        float k0 = kern[0*49+ki], k1 = kern[1*49+ki], k2 = kern[2*49+ki];
        float k3 = kern[3*49+ki], k4 = kern[4*49+ki], k5 = kern[5*49+ki];
        bool in5 = (dy >= 1 && dy <= 5 && dx >= 1 && dx <= 5);
        for (int c = 0; c < 8; ++c) {
          float v = XS(ly + dy, lx + dx, c0 + c);
          acc[c][0] += v * k0; acc[c][1] += v * k1; acc[c][2] += v * k2;
          acc[c][3] += v * k3; acc[c][4] += v * k4; acc[c][5] += v * k5;
          if (in5) m5[c] = fmaxf(m5[c], v);
        }
      }
    for (int c = 0; c < 8; ++c) {
      PA(p, c0 + c) = (_Float16)XS(ly + 3, lx + 3, c0 + c);         // x
      for (int k = 0; k < 6; ++k)
        PA(p, 16 + k * 16 + c0 + c) = (_Float16)acc[c][k];          // conv (k,c)
      PA(p, 112 + c0 + c) = (_Float16)m5[c];                        // maxpool5
    }
    if (!half) {                                                    // pre-life
      float am = -3.0e38f;
      for (int dy = 2; dy <= 4; ++dy)
        for (int dx = 2; dx <= 4; ++dx)
          am = fmaxf(am, XS(ly + dy, lx + dx, 3));
      preL[p] = (am > 0.1f) ? 1 : 0;
    }
  }
  __syncthreads();

  if (tid < NPIX) {
    int ly = tid >> 4, lx = tid & 15;
    preMask[((size_t)b * Hd + ty0 + ly) * Wd + tx0 + lx] = preL[tid];
  }

  // ---- GEMM1 (perc@W0+b0, relu) fused with GEMM2 (h@W1), all WMMA ----
  const int w    = tid >> 5;          // wave id = M-tile = tile row
  const int lane = tid & 31;
  const int lrow = lane & 15, lhi = lane >> 4;

  v16h afrag[4];                      // A: 16 pixels x 128 perc
  for (int ks = 0; ks < 4; ++ks) {
    const _Float16* ap = &PA(w * 16 + lrow, ks * 32 + lhi * 16);
    afrag[ks] = cat8(*(const v8h*)ap, *(const v8h*)(ap + 8));
  }
  v16h w1f[8];                        // W1 B-fragments, preloaded (L2-hot)
  for (int q = 0; q < 8; ++q) {
    const _Float16* wp = W1T + (size_t)lrow * HID + q * 32 + lhi * 16;
    w1f[q] = cat8(*(const v8h*)wp, *(const v8h*)(wp + 8));
  }
  float biasv[16];
  for (int nt = 0; nt < 16; ++nt) biasv[nt] = b0[nt * 16 + lrow];

  v8f dacc = {0.f, 0.f, 0.f, 0.f, 0.f, 0.f, 0.f, 0.f};

  for (int pr = 0; pr < 8; ++pr) {                  // 8 pairs of hidden N-tiles
    for (int sub = 0; sub < 2; ++sub) {
      const int nt = pr * 2 + sub;
      float bias = biasv[nt];
      v8f acc = {bias, bias, bias, bias, bias, bias, bias, bias};
      for (int ks = 0; ks < 4; ++ks) {
        const _Float16* bp = &W0S(nt * 16 + lrow, ks * 32 + lhi * 16);
        v16h bfrag = cat8(*(const v8h*)bp, *(const v8h*)(bp + 8));
        acc = __builtin_amdgcn_wmma_f32_16x16x32_f16(false, afrag[ks], false, bfrag,
                                                     (short)0, acc, false, false);
      }
      for (int r = 0; r < 8; ++r) {                 // relu + D->A relayout
        float hv = fmaxf(acc[r], 0.f);
        STG(w, r + lhi * 8, sub * 16 + lrow) = (_Float16)hv;
      }
    }
    __builtin_amdgcn_wave_barrier();
    asm volatile("s_wait_dscnt 0" ::: "memory");
    const _Float16* hp = &STG(w, lrow, lhi * 16);
    v16h hfrag = cat8(*(const v8h*)hp, *(const v8h*)(hp + 8));
    dacc = __builtin_amdgcn_wmma_f32_16x16x32_f16(false, hfrag, false, w1f[pr],
                                                  (short)0, dacc, false, false);
  }

  // ---- Stochastic residual update, write x_mid ----
  for (int r = 0; r < 8; ++r) {
    int m  = r + lhi * 8;                           // pixel column in tile
    int gy = ty0 + w, gx = tx0 + m;
    size_t pix = ((size_t)b * Hd + gy) * Wd + gx;
    float gate = (stoch[pix] > 0.5f) ? 1.f : 0.f;
    float xold = XS(w + 3, m + 3, lrow);
    xmid[pix * CH + lrow] = xold + dacc[r] * gate;
  }
#undef XS
#undef PA
#undef STG
#undef W0S
}

// ---------------------------------------------------------------------------
__global__ __launch_bounds__(256) void nca_life(
    const float* __restrict__ xmid, const unsigned char* __restrict__ preMask,
    const float* __restrict__ valid, float* __restrict__ xout,
    int Hd, int Wd, int Bn)
{
  size_t idx = (size_t)blockIdx.x * 256 + threadIdx.x;
  size_t total = (size_t)Bn * Hd * Wd;
  if (idx >= total) return;
  int wx = (int)(idx % Wd);
  int hy = (int)((idx / Wd) % Hd);
  float am = -3.0e38f;
  for (int dy = -1; dy <= 1; ++dy) {
    int yy = hy + dy; if (yy < 0 || yy >= Hd) continue;
    for (int dx = -1; dx <= 1; ++dx) {
      int xx = wx + dx; if (xx < 0 || xx >= Wd) continue;
      long long nb = (long long)idx + dy * Wd + dx;
      am = fmaxf(am, xmid[(size_t)nb * CH + 3]);
    }
  }
  bool post = am > 0.1f;
  bool pre  = preMask[idx] != 0;
  float s   = (pre && post) ? valid[idx] : 0.f;
  const float4* src = (const float4*)(xmid + idx * CH);
  float4*       dst = (float4*)(xout + idx * CH);
  for (int j = 0; j < 4; ++j) {
    float4 v = src[j];
    v.x *= s; v.y *= s; v.z *= s; v.w *= s;
    dst[j] = v;
  }
}

// ---------------------------------------------------------------------------
extern "C" void kernel_launch(void* const* d_in, const int* in_sizes, int n_in,
                              void* d_out, int out_size, void* d_ws, size_t ws_size,
                              hipStream_t stream) {
  (void)in_sizes; (void)n_in; (void)out_size; (void)ws_size;
  const int B = 8, H = 256, W = 256, STEPS = 16;

  const float* x0    = (const float*)d_in[0];
  const float* valid = (const float*)d_in[1];
  const float* stoch = (const float*)d_in[2];
  const float* W0    = (const float*)d_in[3];
  const float* b0    = (const float*)d_in[4];
  const float* W1    = (const float*)d_in[5];

  char* ws = (char*)d_ws;
  _Float16*      W0T  = (_Float16*)ws;                              // 64 KB
  _Float16*      W1T  = (_Float16*)(ws + 65536);                    //  8 KB
  unsigned char* mask = (unsigned char*)(ws + 65536 + 8192);        // 512 KB
  float*         mid  = (float*)(ws + 65536 + 8192 + 524288);       // 32 MB

  nca_prep<<<128, 256, 0, stream>>>(W0, W1, W0T, W1T);

  float* xout = (float*)d_out;                 // doubles as ping state buffer
  const size_t npix = (size_t)B * H * W;
  dim3 grid(W / TW, H / THT, B);
  for (int s = 0; s < STEPS; ++s) {
    const float* cur = (s == 0) ? x0 : (const float*)xout;
    nca_update<<<grid, 256, SMEM_BYTES, stream>>>(
        cur, mid, mask, stoch + (size_t)s * npix, b0, W0T, W1T, H, W);
    nca_life<<<(unsigned)((npix + 255) / 256), 256, 0, stream>>>(
        mid, mask, valid, xout, H, W, B);
  }
}
